// AE_spikes_13228499271681
// MI455X (gfx1250) — compile-verified, matrix operand order checked by reading, not run on hardware
//
#include <hip/hip_runtime.h>
#include <stdint.h>

// ---------------------------------------------------------------------------
// SNN autoencoder, fully fused for MI455X (gfx1250, wave32, WMMA).
//   out = ((sum_t s3_t) @ W4^T) / 16 + b4
// Encoder spikes precomputed as per-element 16-bit masks in LDS:
//   s0_t = floor((t+1)f) - floor(t f)   (bit t of mask)
// 8 waves / block (2 per SIMD32) so one wave's WMMA burst hides another
// wave's fire/staging phases.  Persistent f32 potentials live in WMMA
// C-fragment layout (v8f per 16x16 tile) in VGPRs across all 16 timesteps.
// W2/W3 fragment blobs are async-copied (global_load_async_to_lds_b128,
// ASYNCcnt) into LDS once and reused.  All GEMMs: v_wmma_f32_16x16x32_f16.
// ---------------------------------------------------------------------------

typedef __attribute__((ext_vector_type(16))) _Float16 v16h;
typedef __attribute__((ext_vector_type(8)))  _Float16 v8h;
typedef __attribute__((ext_vector_type(8)))  float    v8f;
typedef __attribute__((ext_vector_type(8)))  uint32_t v8u;

#define BATCH    32768
#define IN_SHAPE 784
#define HIDDEN   128
#define DUR      16
#define TILE_B   128         // batch rows per block (8 waves x 16 rows)
#define NTHREADS 256
#define MASK_STRIDE 800      // 784 padded to 25 k-slices of 32

// ---------------------------------------------------------------------------
// Weight pre-swizzle: W is [N][K] row-major f32 (x @ W.T uses B[k][n] = W[n][k]).
// Emit f16 fragments so each lane's 16 values are one contiguous 32B chunk:
//   lane l -> n = nt*16 + (l&15), b0 = (l&16)?8:0
//   slot j -> k = s*32 + b0 + (j<8 ? j : 8+j)
// dst element index = ((s*NT + nt)*32 + lane)*16 + j
// ---------------------------------------------------------------------------
__global__ void prep_swizzle(const float* __restrict__ W,
                             _Float16* __restrict__ dst,
                             int S, int NT, int K) {
    int idx = blockIdx.x * 256 + threadIdx.x;
    int total = S * NT * 512;
    if (idx >= total) return;
    int j    = idx & 15;
    int lane = (idx >> 4) & 31;
    int nt   = (idx >> 9) % NT;
    int s    = idx / (NT << 9);
    int b0   = (lane & 16) ? 8 : 0;
    int n    = nt * 16 + (lane & 15);
    int k    = s * 32 + b0 + ((j < 8) ? j : (8 + j));
    float v  = (k < K) ? W[n * K + k] : 0.0f;
    dst[idx] = (_Float16)v;
}

// A/B fragment from row-major f16: 8 halves at base (K..K+7), 8 at base+16.
__device__ __forceinline__ v16h frag_rowmajor(const _Float16* base) {
    v8h lo = *(const v8h*)(base);
    v8h hi = *(const v8h*)(base + 16);
    v16h a;
#pragma unroll
    for (int j = 0; j < 8; ++j) { a[j] = lo[j]; a[8 + j] = hi[j]; }
    return a;
}

// Pre-swizzled fragment load (32B contiguous per lane) -- global or LDS.
__device__ __forceinline__ v16h frag_swz(const _Float16* fragbase, int lane) {
    const _Float16* p = fragbase + (lane << 4);
    v8h lo = *(const v8h*)(p);
    v8h hi = *(const v8h*)(p + 8);
    v16h b;
#pragma unroll
    for (int j = 0; j < 8; ++j) { b[j] = lo[j]; b[8 + j] = hi[j]; }
    return b;
}

// Branchless spike A-fragment: bit t of each u16 -> f16 1.0/0.0, two packed
// per dword:  ((w>>t) & 0x10001) * 0x3C00.
__device__ __forceinline__ v16h spike_frag(const uint32_t* ql,
                                           const uint32_t* qh, int t) {
    v8u u;
#pragma unroll
    for (int d = 0; d < 4; ++d) {
        u[d]     = ((ql[d] >> t) & 0x10001u) * 0x3C00u;
        u[4 + d] = ((qh[d] >> t) & 0x10001u) * 0x3C00u;
    }
    return __builtin_bit_cast(v16h, u);
}

__device__ __forceinline__ v8f wmma_f16(v16h a, v16h b, v8f c) {
    return __builtin_amdgcn_wmma_f32_16x16x32_f16(false, a, false, b,
                                                  (short)0, c, false, false);
}

// Async global->LDS 16B copy (per-lane), tracked by ASYNCcnt.
__device__ __forceinline__ void async_copy_b128(uint32_t lds_off,
                                                const void* gaddr) {
    asm volatile("global_load_async_to_lds_b128 %0, %1, off"
                 :: "v"(lds_off), "v"((uint64_t)(uintptr_t)gaddr)
                 : "memory");
}

__global__ __launch_bounds__(NTHREADS)
void snn_fused(const float* __restrict__ feat,
               const float* __restrict__ b1, const float* __restrict__ b2,
               const float* __restrict__ b3, const float* __restrict__ b4,
               const _Float16* __restrict__ w1f, const _Float16* __restrict__ w2f,
               const _Float16* __restrict__ w3f, const _Float16* __restrict__ w4f,
               float* __restrict__ out) {
    __shared__ uint16_t sMask[TILE_B * MASK_STRIDE];   // 200 KB spike masks
    __shared__ _Float16 sStg[8][16 * HIDDEN];          // 32 KB staging tiles
    __shared__ _Float16 sW2f[4 * 8 * 512];             // 32 KB W2 fragments
    __shared__ _Float16 sW3f[4 * 8 * 512];             // 32 KB W3 fragments
    __shared__ float sB4[IN_SHAPE];                    // ~3 KB decoder bias

    const int tid  = threadIdx.x;
    const int lane = tid & 31;
    const int wave = tid >> 5;
    const int row0 = blockIdx.x * TILE_B;

    // ---- async-stage W2/W3 fragment blobs into LDS (ASYNCcnt path) ----
    {
        const uint32_t ldsW2 = (uint32_t)(uintptr_t)(&sW2f[0]);
        const uint32_t ldsW3 = (uint32_t)(uintptr_t)(&sW3f[0]);
#pragma unroll 1
        for (int i = tid; i < 2048; i += NTHREADS) {   // 2048 x 16B = 32 KB each
            async_copy_b128(ldsW2 + i * 16, (const char*)w2f + i * 16);
            async_copy_b128(ldsW3 + i * 16, (const char*)w3f + i * 16);
        }
    }

    // ---- decoder bias to LDS; layer biases to registers (loop-invariant) ----
    for (int i = tid; i < IN_SHAPE; i += NTHREADS) sB4[i] = b4[i];
    float bias1[8], bias2[8], bias3[8];
#pragma unroll
    for (int nt = 0; nt < 8; ++nt) {
        const int n = (nt << 4) + (lane & 15);
        bias1[nt] = b1[n];
        bias2[nt] = b2[n];
        bias3[nt] = b3[n];
    }

    // ---- stage encoder spike bitmask (bit t => spike at step t) ----
    for (int idx = tid; idx < TILE_B * MASK_STRIDE; idx += NTHREADS) {
        int r = idx / MASK_STRIDE, c = idx - r * MASK_STRIDE;
        unsigned m = 0;
        if (c < IN_SHAPE) {
            float f = feat[(row0 + r) * IN_SHAPE + c];
            float prev = 0.0f;
#pragma unroll
            for (int t = 1; t <= DUR; ++t) {
                float cur = floorf(f * (float)t);
                m |= ((unsigned)(cur - prev)) << (t - 1);
                prev = cur;
            }
        }
        sMask[idx] = (uint16_t)m;
    }
    asm volatile("s_wait_asynccnt 0" ::: "memory");
    __syncthreads();

    // persistent potentials (f32 C-fragments) + spike counts (f16, exact <=16)
    v8f p1[8] = {}, p2[8] = {}, p3[8] = {};
    v8h cnt[8] = {};

    _Float16* stg = sStg[wave];
    const int b0    = (lane & 16) ? 8 : 0;   // K sub-run offset for A/B frags
    const int mhalf = (lane & 16) ? 8 : 0;   // M offset for C-fragment layout
    const uint16_t* maskrow = &sMask[((wave << 4) + (lane & 15)) * MASK_STRIDE];

#pragma unroll 1
    for (int t = 0; t < DUR; ++t) {
        // ---- Layer 1: p1 += s0 @ W1^T   (K=784 -> 25 slices, N=128 -> 8 tiles)
#pragma unroll 1
        for (int s = 0; s < 25; ++s) {
            const uint16_t* q = maskrow + s * 32 + b0;
            v16h a = spike_frag((const uint32_t*)q, (const uint32_t*)(q + 16), t);
#pragma unroll
            for (int nt = 0; nt < 8; ++nt) {
                v16h bf = frag_swz(w1f + (((s << 3) + nt) << 9), lane);
                p1[nt] = wmma_f16(a, bf, p1[nt]);
            }
        }
        // fire layer 1 -> s1 staged row-major [m][k] in LDS
#pragma unroll
        for (int nt = 0; nt < 8; ++nt) {
            const int n = (nt << 4) + (lane & 15);
#pragma unroll
            for (int j = 0; j < 8; ++j) {
                float p  = p1[nt][j] + bias1[nt];
                float sp = (p >= 1.0f) ? 1.0f : 0.0f;
                p1[nt][j] = p - sp;
                stg[(j + mhalf) * HIDDEN + n] = (_Float16)sp;
            }
        }

        // ---- Layer 2: p2 += s1 @ W2^T   (K=128 -> 4 slices, weights in LDS)
#pragma unroll 1
        for (int s = 0; s < 4; ++s) {
            v16h a = frag_rowmajor(stg + (lane & 15) * HIDDEN + s * 32 + b0);
#pragma unroll
            for (int nt = 0; nt < 8; ++nt) {
                v16h bf = frag_swz(sW2f + (((s << 3) + nt) << 9), lane);
                p2[nt] = wmma_f16(a, bf, p2[nt]);
            }
        }
#pragma unroll
        for (int nt = 0; nt < 8; ++nt) {
            const int n = (nt << 4) + (lane & 15);
#pragma unroll
            for (int j = 0; j < 8; ++j) {
                float p  = p2[nt][j] + bias2[nt];
                float sp = (p >= 1.0f) ? 1.0f : 0.0f;
                p2[nt][j] = p - sp;
                stg[(j + mhalf) * HIDDEN + n] = (_Float16)sp;
            }
        }

        // ---- Layer 3: p3 += s2 @ W3^T ; accumulate spike counts (f16)
#pragma unroll 1
        for (int s = 0; s < 4; ++s) {
            v16h a = frag_rowmajor(stg + (lane & 15) * HIDDEN + s * 32 + b0);
#pragma unroll
            for (int nt = 0; nt < 8; ++nt) {
                v16h bf = frag_swz(sW3f + (((s << 3) + nt) << 9), lane);
                p3[nt] = wmma_f16(a, bf, p3[nt]);
            }
        }
#pragma unroll
        for (int nt = 0; nt < 8; ++nt) {
#pragma unroll
            for (int j = 0; j < 8; ++j) {
                float p  = p3[nt][j] + bias3[nt];
                float sp = (p >= 1.0f) ? 1.0f : 0.0f;
                p3[nt][j] = p - sp;
                cnt[nt][j] += (_Float16)sp;
            }
        }
    } // timesteps

    // ---- Decoder output, collapsed: out = (cnt @ W4^T)/16 + b4
#pragma unroll
    for (int nt = 0; nt < 8; ++nt) {
        const int n = (nt << 4) + (lane & 15);
#pragma unroll
        for (int j = 0; j < 8; ++j)
            stg[(j + mhalf) * HIDDEN + n] = cnt[nt][j];
    }
    v16h a4[4];
#pragma unroll
    for (int s = 0; s < 4; ++s)
        a4[s] = frag_rowmajor(stg + (lane & 15) * HIDDEN + s * 32 + b0);

    const int rowbase = row0 + (wave << 4);
#pragma unroll 1
    for (int nt = 0; nt < 49; ++nt) {
        v8f c = {};
#pragma unroll
        for (int s = 0; s < 4; ++s) {
            v16h bf = frag_swz(w4f + ((s * 49 + nt) << 9), lane);
            c = wmma_f16(a4[s], bf, c);
        }
        const int n = (nt << 4) + (lane & 15);
        const float bias = sB4[n];
#pragma unroll
        for (int j = 0; j < 8; ++j)
            out[(rowbase + j + mhalf) * IN_SHAPE + n] = c[j] * (1.0f / 16.0f) + bias;
    }
}

extern "C" void kernel_launch(void* const* d_in, const int* in_sizes, int n_in,
                              void* d_out, int out_size, void* d_ws, size_t ws_size,
                              hipStream_t stream) {
    const float* feat = (const float*)d_in[0];
    const float* W1 = (const float*)d_in[1]; const float* b1 = (const float*)d_in[2];
    const float* W2 = (const float*)d_in[3]; const float* b2 = (const float*)d_in[4];
    const float* W3 = (const float*)d_in[5]; const float* b3 = (const float*)d_in[6];
    const float* W4 = (const float*)d_in[7]; const float* b4 = (const float*)d_in[8];

    // ws layout (f16 elements): W1 frags | W2 frags | W3 frags | W4 frags
    _Float16* ws  = (_Float16*)d_ws;
    _Float16* w1f = ws;                       // 25*8*512  = 102400 halves
    _Float16* w2f = w1f + 25 * 8 * 512;       // 4*8*512   = 16384
    _Float16* w3f = w2f + 4 * 8 * 512;
    _Float16* w4f = w3f + 4 * 8 * 512;        // 4*49*512  = 100352
    (void)ws_size; (void)in_sizes; (void)n_in; (void)out_size;

    prep_swizzle<<<(25 * 8 * 512 + 255) / 256, 256, 0, stream>>>(W1, w1f, 25, 8, IN_SHAPE);
    prep_swizzle<<<(4 * 8 * 512 + 255) / 256, 256, 0, stream>>>(W2, w2f, 4, 8, HIDDEN);
    prep_swizzle<<<(4 * 8 * 512 + 255) / 256, 256, 0, stream>>>(W3, w3f, 4, 8, HIDDEN);
    prep_swizzle<<<(4 * 49 * 512 + 255) / 256, 256, 0, stream>>>(W4, w4f, 4, 49, HIDDEN);

    snn_fused<<<BATCH / TILE_B, NTHREADS, 0, stream>>>(feat, b1, b2, b3, b4,
                                                       w1f, w2f, w3f, w4f,
                                                       (float*)d_out);
}
